// FullModel_10075993277104
// MI455X (gfx1250) — matrix operand order repaired and checked by reading, not compiled, not run on hardware
//
#include <hip/hip_runtime.h>

// ---------------------------------------------------------------------------
// CDNA5 (gfx1250) implementation. All matrix contractions use
// v_wmma_f32_16x16x32_bf16 (wave32). GEMM operands that are reused are
// pre-packed into fragment-native layouts so each fragment load is a pair of
// contiguous b128 loads per lane; index math is 32-bit.
// ---------------------------------------------------------------------------

typedef __attribute__((ext_vector_type(16))) __bf16 bf16x16;
typedef __attribute__((ext_vector_type(8)))  __bf16 bf16x8;
typedef __attribute__((ext_vector_type(8)))  float  f32x8;

__device__ __forceinline__ f32x8 wmma_bf16(bf16x16 a, bf16x16 b, f32x8 c) {
  return __builtin_amdgcn_wmma_f32_16x16x32_bf16(false, a, false, b, (short)0, c, false, false);
}

// Load one packed fragment (32 lanes x 16 bf16, 32B/lane contiguous).
__device__ __forceinline__ bf16x16 frag_load(const __bf16* __restrict__ P, int fragIdx) {
  return *(const bf16x16*)(P + ((fragIdx * 32 + (int)(threadIdx.x & 31)) << 4));
}
// A fragment (16x32) from bf16 row-major [.., ld]; 16 contiguous per lane.
__device__ __forceinline__ bf16x16 row_a_load(const __bf16* __restrict__ X, int row_base,
                                              int ld, int kbase) {
  int lane = threadIdx.x & 31;
  const __bf16* p = X + (row_base + (lane & 15)) * ld + kbase + ((lane >> 4) << 4);
  bf16x8 lo = *(const bf16x8*)p;
  bf16x8 hi = *(const bf16x8*)(p + 8);
  bf16x16 a;
#pragma unroll
  for (int e = 0; e < 8; ++e) { a[e] = lo[e]; a[e + 8] = hi[e]; }
  return a;
}
// A fragment from f32 row-major, converting to bf16 (4x float4 per lane).
__device__ __forceinline__ bf16x16 row_a_load_f32(const float* __restrict__ X, int row_base,
                                                  int ld, int kbase) {
  int lane = threadIdx.x & 31;
  const float* p = X + (row_base + (lane & 15)) * ld + kbase + ((lane >> 4) << 4);
  float4 q0 = *(const float4*)p;
  float4 q1 = *(const float4*)(p + 4);
  float4 q2 = *(const float4*)(p + 8);
  float4 q3 = *(const float4*)(p + 12);
  bf16x16 a;
  a[0] = (__bf16)q0.x;  a[1] = (__bf16)q0.y;  a[2] = (__bf16)q0.z;  a[3] = (__bf16)q0.w;
  a[4] = (__bf16)q1.x;  a[5] = (__bf16)q1.y;  a[6] = (__bf16)q1.z;  a[7] = (__bf16)q1.w;
  a[8] = (__bf16)q2.x;  a[9] = (__bf16)q2.y;  a[10] = (__bf16)q2.z; a[11] = (__bf16)q2.w;
  a[12] = (__bf16)q3.x; a[13] = (__bf16)q3.y; a[14] = (__bf16)q3.z; a[15] = (__bf16)q3.w;
  return a;
}

// ----------------------------- pack kernels --------------------------------
// Pack f32 W[K,N] (N mult of 16, K mult of 32) into B-fragment layout:
// P[((ks*(N/16)+jt)*32+lane)*16+e] = W[k,c], k = ks*32 + ((lane>>4)<<4)+e,
// c = jt*16 + (lane&15). Rows k >= Kvalid are zero.
__global__ void pack_w_kernel(const float* __restrict__ W, __bf16* __restrict__ P,
                              int K, int N, int Kvalid) {
  int idx = blockIdx.x * blockDim.x + threadIdx.x;
  if (idx >= K * N) return;
  int e = idx & 15, lane = (idx >> 4) & 31, f = idx >> 9;
  int nt = N >> 4;
  int jt = f % nt, ks = f / nt;
  int k = ks * 32 + ((lane >> 4) << 4) + e;
  int c = jt * 16 + (lane & 15);
  P[idx] = (k < Kvalid) ? (__bf16)W[k * N + c] : (__bf16)0.f;
}

// Row-normalize adjacency -> zero-padded bf16 row-major [2][336][352]
__global__ void adj_prep_kernel(const float* __restrict__ adjs, __bf16* __restrict__ adj_bf) {
  int row = blockIdx.x;  // 0 .. 2*336-1
  int a = row / 336, r = row % 336;
  __bf16* dst = adj_bf + (a * 336 + r) * 352;
  if (r >= 325) {
    for (int c = threadIdx.x; c < 352; c += blockDim.x) dst[c] = (__bf16)0.f;
    return;
  }
  const float* src = adjs + (a * 325 + r) * 325;
  __shared__ float red[128];
  float s = 0.f;
  for (int c = threadIdx.x; c < 325; c += blockDim.x) s += src[c];
  red[threadIdx.x] = s;
  __syncthreads();
  for (int off = 64; off > 0; off >>= 1) {
    if (threadIdx.x < off) red[threadIdx.x] += red[threadIdx.x + off];
    __syncthreads();
  }
  float inv = 1.f / fmaxf(red[0], 1e-6f);
  for (int c = threadIdx.x; c < 352; c += blockDim.x)
    dst[c] = (c < 325) ? (__bf16)(src[c] * inv) : (__bf16)0.f;
}
// Pack adjacency into A-fragment layout: adjp[a][mt(21)][ks(11)][lane][e]
__global__ void adj_pack_kernel(const __bf16* __restrict__ adj_bf, __bf16* __restrict__ adjp) {
  int idx = blockIdx.x * blockDim.x + threadIdx.x;
  if (idx >= 2 * 21 * 11 * 32 * 16) return;
  int e = idx & 15, lane = (idx >> 4) & 31, f = idx >> 9;
  int ks = f % 11, rest = f / 11;
  int mt = rest % 21, a = rest / 21;
  int m = mt * 16 + (lane & 15);
  int klo = (lane >> 4) << 4;
  int kin = klo + e;
  adjp[idx] = adj_bf[(a * 336 + m) * 352 + ks * 32 + kin];
}

// ----------------------------- small utility kernels -----------------------
__global__ void copy_f32_kernel(const float* __restrict__ s, float* __restrict__ d, int n) {
  int i = blockIdx.x * blockDim.x + threadIdx.x;
  if (i < n) d[i] = s[i];
}
__global__ void max_kernel(float* __restrict__ a, const float* __restrict__ b, int n) {
  int i = blockIdx.x * blockDim.x + threadIdx.x;
  if (i < n) a[i] = fmaxf(a[i], b[i]);
}
__global__ void avg_kernel(const float* __restrict__ a, const float* __restrict__ b,
                           float* __restrict__ o, int n) {
  int i = blockIdx.x * blockDim.x + threadIdx.x;
  if (i < n) o[i] = 0.5f * (a[i] + b[i]);
}

// x = relu(feat*in_w+in_b) + time_emb[tod] + (dow @ date_w + date_b) + node_emb
__global__ void embed_kernel(const float* __restrict__ feat, const int* __restrict__ tod,
                             const float* __restrict__ dow, const float* __restrict__ in_w,
                             const float* __restrict__ in_b, const float* __restrict__ time_emb,
                             const float* __restrict__ date_w, const float* __restrict__ date_b,
                             const float* __restrict__ node_emb, float* __restrict__ out) {
  int idx = blockIdx.x * blockDim.x + threadIdx.x;
  const int TOT = 64 * 12 * 325 * 64;
  if (idx >= TOT) return;
  int h = idx & 63;
  int r = idx >> 6;  // (b*12+t)*325+n
  int n = r % 325;
  int bt = r / 325;
  float ie = fmaxf(feat[r] * in_w[h] + in_b[h], 0.f);
  float te = time_emb[tod[bt] * 64 + h];
  float de = date_b[h];
  const float* dwr = dow + bt * 7;
#pragma unroll
  for (int d = 0; d < 7; ++d) de += dwr[d] * date_w[d * 64 + h];
  out[idx] = ie + te + de + node_emb[n * 64 + h];
}

// LayerNorm(H=64) in-place, one wave per row (2 values per lane)
__global__ void ln_kernel(float* __restrict__ x, const float* __restrict__ sc,
                          const float* __restrict__ bi, int rows) {
  int wid = (blockIdx.x * blockDim.x + threadIdx.x) >> 5;
  int lane = threadIdx.x & 31;
  if (wid >= rows) return;
  float* row = x + wid * 64;
  float v0 = row[lane], v1 = row[lane + 32];
  float s = v0 + v1;
#pragma unroll
  for (int m = 1; m < 32; m <<= 1) s += __shfl_xor(s, m, 32);
  float mu = s * (1.f / 64.f);
  float d0 = v0 - mu, d1 = v1 - mu;
  float q = d0 * d0 + d1 * d1;
#pragma unroll
  for (int m = 1; m < 32; m <<= 1) q += __shfl_xor(q, m, 32);
  float inv = rsqrtf(q * (1.f / 64.f) + 1e-5f);
  row[lane] = d0 * inv * sc[lane] + bi[lane];
  row[lane + 32] = d1 * inv * sc[lane + 32] + bi[lane + 32];
}

// ---- backbone dual linear: writes L1,L2 directly in the adjacency GEMM's
// packed B-fragment layout Lp[bt][ks(11)][jt(4)][lane][e], zero-filling K-pad.
__global__ void dual_lin_bb_kernel(const float* __restrict__ X, const __bf16* __restrict__ w1p,
                                   const __bf16* __restrict__ w2p, const float* __restrict__ b1,
                                   const float* __restrict__ b2, __bf16* __restrict__ L1p,
                                   __bf16* __restrict__ L2p) {
  int wid = (blockIdx.x * blockDim.x + threadIdx.x) >> 5;
  if (wid >= 768 * 21 * 4) return;
  int jt = wid & 3;
  int t = wid >> 2;
  int rt = t % 21, bt = t / 21;
  int rowbase = bt * 325 + rt * 16;  // rows >=325 in-tile are garbage, masked on store
  f32x8 acc1 = {}, acc2 = {};
#pragma unroll
  for (int ks = 0; ks < 2; ++ks) {
    bf16x16 a = row_a_load_f32(X, rowbase, 64, ks * 32);
    acc1 = wmma_bf16(a, frag_load(w1p, ks * 4 + jt), acc1);
    acc2 = wmma_bf16(a, frag_load(w2p, ks * 4 + jt), acc2);
  }
  int lane = threadIdx.x & 31;
  int c = lane & 15, hi = lane >> 4;
  int h = jt * 16 + c;
  float bias1 = b1[h], bias2 = b2[h];
  int ks = rt >> 1, half = rt & 1;
  int fragLane = ((bt * 11 + ks) * 4 + jt) * 32 + (c + 16 * half);
  int e0 = hi * 8;
  int nbase = rt * 16 + e0;
  bf16x8 v1, v2;
#pragma unroll
  for (int i = 0; i < 8; ++i) {
    bool ok = (nbase + i) < 325;
    v1[i] = ok ? (__bf16)(acc1[i] + bias1) : (__bf16)0.f;
    v2[i] = ok ? (__bf16)(acc2[i] + bias2) : (__bf16)0.f;
  }
  *(bf16x8*)(L1p + fragLane * 16 + e0) = v1;
  *(bf16x8*)(L2p + fragLane * 16 + e0) = v2;
}

// heads dual linear: f32 outputs contiguous [R,64]
__global__ void dual_lin_f32_kernel(const float* __restrict__ X, const __bf16* __restrict__ w1p,
                                    const __bf16* __restrict__ w2p, const float* __restrict__ b1,
                                    const float* __restrict__ b2, float* __restrict__ o1,
                                    float* __restrict__ o2, int rtiles) {
  int wid = (blockIdx.x * blockDim.x + threadIdx.x) >> 5;
  if (wid >= rtiles * 4) return;
  int jt = wid & 3;
  int rt = wid >> 2;
  f32x8 acc1 = {}, acc2 = {};
#pragma unroll
  for (int ks = 0; ks < 2; ++ks) {
    bf16x16 a = row_a_load_f32(X, rt * 16, 64, ks * 32);
    acc1 = wmma_bf16(a, frag_load(w1p, ks * 4 + jt), acc1);
    acc2 = wmma_bf16(a, frag_load(w2p, ks * 4 + jt), acc2);
  }
  int lane = threadIdx.x & 31;
  int h = jt * 16 + (lane & 15);
  int mh = (lane >> 4) << 3;
  float bias1 = b1[h], bias2 = b2[h];
#pragma unroll
  for (int r = 0; r < 8; ++r) {
    int row = rt * 16 + r + mh;
    o1[row * 64 + h] = acc1[r] + bias1;
    o2[row * 64 + h] = acc2[r] + bias2;
  }
}

// g1 = adj@L1, g2 = adj@L2 per (b,t); dst = relu(g1*g2)+g1 + x (pre-LN)
__global__ void adj_gemm_kernel(const __bf16* __restrict__ adjp, const __bf16* __restrict__ L1p,
                                const __bf16* __restrict__ L2p, const float* __restrict__ xres,
                                float* __restrict__ dst) {
  int wid = (blockIdx.x * blockDim.x + threadIdx.x) >> 5;
  if (wid >= 768 * 84) return;
  int bt = wid / 84;
  int tt = wid % 84;
  int mt = tt >> 2, jt = tt & 3;
  f32x8 a1 = {}, a2 = {};
#pragma unroll 1
  for (int ks = 0; ks < 11; ++ks) {
    bf16x16 af = frag_load(adjp, mt * 11 + ks);
    int bfrag = (bt * 11 + ks) * 4 + jt;
    a1 = wmma_bf16(af, frag_load(L1p, bfrag), a1);
    a2 = wmma_bf16(af, frag_load(L2p, bfrag), a2);
  }
  int lane = threadIdx.x & 31;
  int h = jt * 16 + (lane & 15);
  int mh = (lane >> 4) << 3;
#pragma unroll
  for (int r = 0; r < 8; ++r) {
    int m = mt * 16 + r + mh;
    if (m < 325) {
      int o = (bt * 325 + m) * 64 + h;
      float g1 = a1[r], g2 = a2[r];
      dst[o] = fmaxf(g1 * g2, 0.f) + g1 + xres[o];
    }
  }
}

// STIGCN elementwise: g = 0.5*(L[t-1]+L[t]) (t==0 uses bias); pre-LN out
__global__ void stig_ew_kernel(const float* __restrict__ L1, const float* __restrict__ L2,
                               const float* __restrict__ x, const float* __restrict__ b1,
                               const float* __restrict__ b2, float* __restrict__ dst,
                               int Tk, int tot) {
  int idx = blockIdx.x * blockDim.x + threadIdx.x;
  if (idx >= tot) return;
  int h = idx & 63;
  int r = idx >> 6;
  int t = (r / 325) % Tk;
  const int ts = 325 * 64;
  float l1 = L1[idx], l2 = L2[idx];
  float p1 = (t > 0) ? L1[idx - ts] : b1[h];
  float p2 = (t > 0) ? L2[idx - ts] : b2[h];
  float g1 = 0.5f * (p1 + l1), g2 = 0.5f * (p2 + l2);
  dst[idx] = fmaxf(g1 * g2, 0.f) + g1 + x[idx];
}

// temporal mean pooling from [B,12,N,H] -> [B,Tp,N,H]
__global__ void pool_kernel(const float* __restrict__ x, float* __restrict__ y,
                            int Tp, int r, int tot) {
  int idx = blockIdx.x * blockDim.x + threadIdx.x;
  if (idx >= tot) return;
  int h = idx & 63;
  int row = idx >> 6;
  int n = row % 325;
  int btp = row / 325;
  int tp = btp % Tp;
  int b = btp / Tp;
  float s = 0.f;
  for (int i = 0; i < r; ++i)
    s += x[((b * 12 + tp * r + i) * 325 + n) * 64 + h];
  y[idx] = s / (float)r;
}

// x [B,M,64] f32 -> row-major bf16 [B,Mpad,64] (pad rows zero) for assign A
__global__ void cvt_feat_kernel(const float* __restrict__ x, __bf16* __restrict__ f,
                                int M, int Mpad) {
  int idx = blockIdx.x * blockDim.x + threadIdx.x;
  int tot = 64 * Mpad * 64;
  if (idx >= tot) return;
  int h = idx & 63;
  int row = idx >> 6;
  int m = row % Mpad;
  int b = row / Mpad;
  f[idx] = (m < M) ? (__bf16)x[(b * M + m) * 64 + h] : (__bf16)0.f;
}
// x [B,M,64] f32 -> packed B-fragment layout featp[b][ks][jt(4)][lane][e] for hf
__global__ void pack_feat_kernel(const float* __restrict__ x, __bf16* __restrict__ P,
                                 int M, int Mpad) {
  int idx = blockIdx.x * blockDim.x + threadIdx.x;
  int tot = 64 * Mpad * 64;
  if (idx >= tot) return;
  int e = idx & 15, lane = (idx >> 4) & 31, f = idx >> 9;
  int jt = f & 3, f2 = f >> 2;
  int ksn = Mpad >> 5;
  int ks = f2 % ksn, b = f2 / ksn;
  int m = ks * 32 + ((lane >> 4) << 4) + e;
  int h = jt * 16 + (lane & 15);
  P[idx] = (m < M) ? (__bf16)x[(b * M + m) * 64 + h] : (__bf16)0.f;
}

// assign = softmax(feat @ edge_clf); writes row-major [B,Mpad,32] and
// transposed [B,32,Mpad] (both zero for pad rows)
__global__ void assign_kernel(const __bf16* __restrict__ feat, const __bf16* __restrict__ eclfp,
                              __bf16* __restrict__ assign, __bf16* __restrict__ assignT,
                              int M, int Mpad) {
  int wid = (blockIdx.x * blockDim.x + threadIdx.x) >> 5;
  int mtb = Mpad >> 4;
  if (wid >= 64 * mtb) return;
  int b = wid / mtb;
  int mt = wid % mtb;
  f32x8 a0 = {}, a1 = {};
#pragma unroll
  for (int ks = 0; ks < 2; ++ks) {
    bf16x16 af = row_a_load(feat, b * Mpad + mt * 16, 64, ks * 32);
    a0 = wmma_bf16(af, frag_load(eclfp, ks * 2 + 0), a0);
    a1 = wmma_bf16(af, frag_load(eclfp, ks * 2 + 1), a1);
  }
  int lane = threadIdx.x & 31;
  int c = lane & 15;
  int mh = (lane >> 4) << 3;
#pragma unroll
  for (int r = 0; r < 8; ++r) {
    float v0 = a0[r], v1 = a1[r];
    float mx = fmaxf(v0, v1);
#pragma unroll
    for (int o = 1; o < 16; o <<= 1) mx = fmaxf(mx, __shfl_xor(mx, o, 32));
    float e0v = __expf(v0 - mx), e1v = __expf(v1 - mx);
    float s = e0v + e1v;
#pragma unroll
    for (int o = 1; o < 16; o <<= 1) s += __shfl_xor(s, o, 32);
    float inv = 1.f / s;
    int m = mt * 16 + r + mh;
    __bf16 w0 = (m < M) ? (__bf16)(e0v * inv) : (__bf16)0.f;
    __bf16 w1 = (m < M) ? (__bf16)(e1v * inv) : (__bf16)0.f;
    int o = (b * Mpad + m) * 32 + c;
    assign[o] = w0;
    assign[o + 16] = w1;
    int oT = (b * 32 + c) * Mpad + m;
    assignT[oT] = w0;
    assignT[oT + 16 * Mpad] = w1;
  }
}

// hf[b,e,h] = sum_m assign[b,m,e] * feat[b,m,h]  (A rows from assignT)
__global__ void hf_kernel(const __bf16* __restrict__ assignT, const __bf16* __restrict__ featp,
                          float* __restrict__ hf, int Mpad) {
  int wid = (blockIdx.x * blockDim.x + threadIdx.x) >> 5;
  if (wid >= 64 * 8) return;
  int b = wid >> 3;
  int tt = wid & 7;
  int et = tt >> 2, ht = tt & 3;
  const __bf16* fp = featp + b * Mpad * 64;
  f32x8 acc = {};
  int ksn = Mpad >> 5;
#pragma unroll 1
  for (int ks = 0; ks < ksn; ++ks) {
    bf16x16 af = row_a_load(assignT, b * 32 + et * 16, Mpad, ks * 32);
    acc = wmma_bf16(af, frag_load(fp, ks * 4 + ht), acc);
  }
  int lane = threadIdx.x & 31;
  int h = ht * 16 + (lane & 15);
  int mh = (lane >> 4) << 3;
#pragma unroll
  for (int r = 0; r < 8; ++r) {
    int e = et * 16 + r + mh;
    hf[(b * 32 + e) * 64 + h] = acc[r];
  }
}

// out = relu(edge_map @ hf) + hf, written pre-packed as the scatter GEMM's B
__global__ void edge_mix_kernel(const float* __restrict__ hf, const float* __restrict__ emap,
                                __bf16* __restrict__ outp) {
  int b = blockIdx.x;
  __shared__ float hs[32 * 64];
  for (int i = threadIdx.x; i < 2048; i += blockDim.x) hs[i] = hf[b * 2048 + i];
  __syncthreads();
  for (int pidx = threadIdx.x; pidx < 2048; pidx += blockDim.x) {
    int ee = pidx & 15, lane = (pidx >> 4) & 31, jt = pidx >> 9;
    int e = ((lane >> 4) << 4) + ee;       // edge row (k of scatter GEMM)
    int h = jt * 16 + (lane & 15);
    float acc = 0.f;
#pragma unroll
    for (int f = 0; f < 32; ++f) acc += emap[e * 32 + f] * hs[f * 64 + h];
    outp[b * 2048 + pidx] = (__bf16)(fmaxf(acc, 0.f) + hs[e * 64 + h]);
  }
}

// y = relu(assign @ out) + x  (pre-LN)
__global__ void scatter_kernel(const __bf16* __restrict__ assign, const __bf16* __restrict__ outp,
                               const float* __restrict__ x, float* __restrict__ dst,
                               int M, int Mpad) {
  int wid = (blockIdx.x * blockDim.x + threadIdx.x) >> 5;
  int mtb = Mpad >> 4;
  if (wid >= 64 * mtb * 4) return;
  int jt = wid & 3;
  int t2 = wid >> 2;
  int b = t2 / mtb;
  int mt = t2 % mtb;
  f32x8 acc = {};
  bf16x16 af = row_a_load(assign, b * Mpad + mt * 16, 32, 0);
  acc = wmma_bf16(af, frag_load(outp + b * 2048, jt), acc);
  int lane = threadIdx.x & 31;
  int h = jt * 16 + (lane & 15);
  int mh = (lane >> 4) << 3;
#pragma unroll
  for (int r = 0; r < 8; ++r) {
    int m = mt * 16 + r + mh;
    if (m < M) {
      int o = (b * M + m) * 64 + h;
      dst[o] = fmaxf(acc[r], 0.f) + x[o];
    }
  }
}

// locals/globals slice into bf16 concat buffers [B,N,384] at col k*64
__global__ void copy_lg_kernel(const float* __restrict__ x, __bf16* __restrict__ lc,
                               __bf16* __restrict__ gc, int Tk, int k) {
  int idx = blockIdx.x * blockDim.x + threadIdx.x;
  const int TOT = 64 * 325 * 64;
  if (idx >= TOT) return;
  int h = idx & 63;
  int row = idx >> 6;
  int n = row % 325;
  int b = row / 325;
  const float* xb = x + ((b * Tk) * 325 + n) * 64 + h;
  const int ts = 325 * 64;
  float s = 0.f;
  for (int t = 0; t < Tk; ++t) s += xb[t * ts];
  int o = (b * 325 + n) * 384 + k * 64 + h;
  lc[o] = (__bf16)xb[(Tk - 1) * ts];
  gc[o] = (__bf16)(s / (float)Tk);
}

// generic WMMA linear: out[R,64] = act(X[R,K] @ Wp + bias); K mult of 32
__global__ void lin_bf16_kernel(const __bf16* __restrict__ X, const __bf16* __restrict__ Wp,
                                const float* __restrict__ bias, float* __restrict__ out,
                                int rtiles, int K, int relu) {
  int wid = (blockIdx.x * blockDim.x + threadIdx.x) >> 5;
  if (wid >= rtiles * 4) return;
  int jt = wid & 3;
  int rt = wid >> 2;
  f32x8 acc = {};
#pragma unroll 1
  for (int ks = 0; ks < K / 32; ++ks) {
    bf16x16 a = row_a_load(X, rt * 16, K, ks * 32);
    acc = wmma_bf16(a, frag_load(Wp, ks * 4 + jt), acc);
  }
  int lane = threadIdx.x & 31;
  int h = jt * 16 + (lane & 15);
  int mh = (lane >> 4) << 3;
  float bi = bias[h];
#pragma unroll
  for (int r = 0; r < 8; ++r) {
    int row = rt * 16 + r + mh;
    float v = acc[r] + bi;
    if (relu) v = fmaxf(v, 0.f);
    out[row * 64 + h] = v;
  }
}

// z = [lf | gf | fut | 0pad] bf16 [B*N,160]
__global__ void build_z_kernel(const float* __restrict__ lf, const float* __restrict__ gf,
                               const float* __restrict__ target, __bf16* __restrict__ z) {
  int idx = blockIdx.x * blockDim.x + threadIdx.x;
  const int TOT = 64 * 325 * 160;
  if (idx >= TOT) return;
  int c = idx % 160;
  int row = idx / 160;
  int n = row % 325;
  int b = row / 325;
  float v;
  if (c < 64) v = lf[row * 64 + c];
  else if (c < 128) v = gf[row * 64 + (c - 64)];
  else if (c < 140) v = target[(b * 12 + (c - 128)) * 325 + n];
  else v = 0.f;
  z[idx] = (__bf16)v;
}

// pred = h1 @ p2_w + p2_b, stored transposed into d_out [B,PRE,N,1]
__global__ void pred_kernel(const float* __restrict__ h1, const float* __restrict__ w,
                            const float* __restrict__ bias, float* __restrict__ out) {
  int idx = blockIdx.x * blockDim.x + threadIdx.x;
  const int TOT = 64 * 325 * 12;
  if (idx >= TOT) return;
  int p = idx % 12;
  int row = idx / 12;
  int n = row % 325;
  int b = row / 325;
  float acc = bias[p];
  const float* hr = h1 + row * 64;
#pragma unroll
  for (int h = 0; h < 64; ++h) acc += hr[h] * w[h * 12 + p];
  out[(b * 12 + p) * 325 + n] = acc;
}

// ---------------------------------------------------------------------------
extern "C" void kernel_launch(void* const* d_in, const int* in_sizes, int n_in,
                              void* d_out, int out_size, void* d_ws, size_t ws_size,
                              hipStream_t stream) {
  (void)in_sizes; (void)n_in; (void)out_size; (void)ws_size;
  const float* feat   = (const float*)d_in[0];
  const int*   tod    = (const int*)d_in[1];
  const float* dow    = (const float*)d_in[2];
  const float* target = (const float*)d_in[3];
  const float* adjs   = (const float*)d_in[4];
  // jax pytree order (sorted dict keys)
  enum { F_B1 = 0, F_B2 = 1, F_LNB = 2, F_LNS = 3, F_W1 = 4, F_W2 = 5 };
  auto bbI = [](int i, int l, int f) { return 5 + (i * 7 + l) * 6 + f; };
  auto hdI = [](int k, int d, int f) { return 93 + (k * 3 + d) * 6 + f; };
  const float* date_b   = (const float*)d_in[89];
  const float* date_w   = (const float*)d_in[90];
  const float* global_b = (const float*)d_in[91];
  const float* global_w = (const float*)d_in[92];
  const float* eclf     = (const float*)d_in[201];
  const float* emap     = (const float*)d_in[202];
  const float* hln_b    = (const float*)d_in[203];
  const float* hln_s    = (const float*)d_in[204];
  const float* in_b     = (const float*)d_in[205];
  const float* in_w     = (const float*)d_in[206];
  const float* local_b  = (const float*)d_in[207];
  const float* local_w  = (const float*)d_in[208];
  const float* node_emb = (const float*)d_in[209];
  const float* p1_b     = (const float*)d_in[210];
  const float* p1_w     = (const float*)d_in[211];
  const float* p2_b     = (const float*)d_in[212];
  const float* p2_w     = (const float*)d_in[213];
  const float* time_emb = (const float*)d_in[214];

  const long S    = 64L * 12 * 325 * 64;  // 15,974,400 activation elems
  const long SPAD = 768L * 352 * 64;      // 17,301,504 packed bf16 elems
  size_t off = 0;
  char* ws = (char*)d_ws;
  auto carve = [&](size_t bytes) { size_t o = off; off += (bytes + 255) & ~(size_t)255; return o; };
  __bf16* adj_bf = (__bf16*)(ws + carve(2L * 336 * 352 * 2));   // row-major padded
  __bf16* adjp   = (__bf16*)(ws + carve(2L * 336 * 352 * 2));   // A-fragment packed
  __bf16* wbf    = (__bf16*)(ws + carve(323584L * 2));          // packed weight pool
  // f32 activation buffers (+32 slack rows for tile-overreach at the tail)
  float* buf0 = (float*)(ws + carve(S * 4 + 8192));
  float* buf1 = (float*)(ws + carve(S * 4 + 8192));
  float* buf2 = (float*)(ws + carve(S * 4 + 8192));
  float* buf3 = (float*)(ws + carve(S * 4 + 8192));
  float* buf4 = (float*)(ws + carve(S * 4 + 8192));
  __bf16* L1p  = (__bf16*)(ws + carve(SPAD * 2));  // reused as featbf (row-major) in heads
  __bf16* L2p  = (__bf16*)(ws + carve(SPAD * 2));  // reused as assign (row-major) in heads
  __bf16* featp   = (__bf16*)(ws + carve(64L * 3904 * 64 * 2));  // packed feat for hf
  __bf16* assignT = (__bf16*)(ws + carve(64L * 3904 * 32 * 2));  // transposed assign
  float*  hfb   = (float*)(ws + carve(64L * 32 * 64 * 4));
  __bf16* outp  = (__bf16*)(ws + carve(64L * 32 * 64 * 2));
  __bf16* lconc = (__bf16*)(ws + carve(64L * 325 * 384 * 2));
  __bf16* gconc = (__bf16*)(ws + carve(64L * 325 * 384 * 2));
  float*  lfb   = (float*)(ws + carve(64L * 325 * 64 * 4));
  float*  gfb   = (float*)(ws + carve(64L * 325 * 64 * 4));
  float*  h1b   = (float*)(ws + carve(64L * 325 * 64 * 4));
  __bf16* zbf   = (__bf16*)(ws + carve(64L * 325 * 160 * 2));

  auto gr = [](long n) { return (unsigned)((n + 255) / 256); };
  auto wblk = [](long waves) { return (unsigned)((waves * 32 + 255) / 256); };

  // ---- pack all weights into fragment layout ----
  auto packw = [&](const float* s, __bf16* d, int K, int N, int Kv) {
    pack_w_kernel<<<gr((long)K * N), 256, 0, stream>>>(s, d, K, N, Kv);
  };
  for (int i = 0; i < 2; ++i)
    for (int l = 0; l < 7; ++l) {
      int j = (i * 7 + l) * 2;
      packw((const float*)d_in[bbI(i, l, F_W1)], wbf + j * 4096, 64, 64, 64);
      packw((const float*)d_in[bbI(i, l, F_W2)], wbf + (j + 1) * 4096, 64, 64, 64);
    }
  for (int k = 0; k < 6; ++k)
    for (int d = 0; d < 3; ++d) {
      int j = 28 + (k * 3 + d) * 2;
      packw((const float*)d_in[hdI(k, d, F_W1)], wbf + j * 4096, 64, 64, 64);
      packw((const float*)d_in[hdI(k, d, F_W2)], wbf + (j + 1) * 4096, 64, 64, 64);
    }
  __bf16* eclfp = wbf + 64 * 4096;
  packw(eclf, eclfp, 64, 32, 64);
  __bf16* localp = eclfp + 2048;
  packw(local_w, localp, 384, 64, 384);
  __bf16* globalp = localp + 24576;
  packw(global_w, globalp, 384, 64, 384);
  __bf16* p1p = globalp + 24576;
  packw(p1_w, p1p, 160, 64, 140);  // zero-pad K 140->160

  // ---- adjacency prep + fragment pack + embedding ----
  adj_prep_kernel<<<2 * 336, 128, 0, stream>>>(adjs, adj_bf);
  adj_pack_kernel<<<gr(2L * 336 * 352), 256, 0, stream>>>(adj_bf, adjp);
  embed_kernel<<<gr(S), 256, 0, stream>>>(feat, tod, dow, in_w, in_b, time_emb,
                                          date_w, date_b, node_emb, buf0);

  // ---- backbone: 2 branches x 7 GNN layers ----
  for (int br = 0; br < 2; ++br) {
    const float* src = buf0;
    float* pp[2] = { buf1, buf2 };
    for (int l = 0; l < 7; ++l) {
      float* dst = pp[l & 1];
      int j = (br * 7 + l) * 2;
      dual_lin_bb_kernel<<<wblk(768L * 21 * 4), 256, 0, stream>>>(
          src, wbf + j * 4096, wbf + (j + 1) * 4096,
          (const float*)d_in[bbI(br, l, F_B1)], (const float*)d_in[bbI(br, l, F_B2)], L1p, L2p);
      adj_gemm_kernel<<<wblk(768L * 84), 256, 0, stream>>>(
          adjp + br * 21 * 11 * 512, L1p, L2p, src, dst);
      ln_kernel<<<wblk(S / 64), 256, 0, stream>>>(
          dst, (const float*)d_in[bbI(br, l, F_LNS)], (const float*)d_in[bbI(br, l, F_LNB)],
          (int)(S / 64));
      src = dst;
    }
    if (br == 0)
      copy_f32_kernel<<<gr(S), 256, 0, stream>>>(buf1, buf3, (int)S);
    else
      max_kernel<<<gr(S), 256, 0, stream>>>(buf3, buf1, (int)S);
  }

  // ---- 6 temporal heads x depth 3 of 0.5*(stigcn + hyper) ----
  const int TKS[6] = { 12, 6, 4, 3, 2, 1 };
  const int RR[6]  = { 1, 2, 3, 4, 6, 12 };
  for (int k = 0; k < 6; ++k) {
    int Tk = TKS[k];
    int M = Tk * 325;
    int Mpad = ((M + 31) / 32) * 32;
    int Rk = 64 * M;
    int EL = Rk * 64;
    pool_kernel<<<gr(EL), 256, 0, stream>>>(buf3, buf0, Tk, RR[k], EL);
    for (int d = 0; d < 3; ++d) {
      int j = 28 + (k * 3 + d) * 2;
      const float* b1 = (const float*)d_in[hdI(k, d, F_B1)];
      const float* b2 = (const float*)d_in[hdI(k, d, F_B2)];
      // STIGCN branch -> buf4
      dual_lin_f32_kernel<<<wblk((long)(Rk / 16) * 4), 256, 0, stream>>>(
          buf0, wbf + j * 4096, wbf + (j + 1) * 4096, b1, b2, buf1, buf2, Rk / 16);
      stig_ew_kernel<<<gr(EL), 256, 0, stream>>>(buf1, buf2, buf0, b1, b2, buf4, Tk, EL);
      ln_kernel<<<wblk(Rk), 256, 0, stream>>>(
          buf4, (const float*)d_in[hdI(k, d, F_LNS)], (const float*)d_in[hdI(k, d, F_LNB)], Rk);
      // Hypergraph branch -> buf1
      cvt_feat_kernel<<<gr(64L * Mpad * 64), 256, 0, stream>>>(buf0, L1p, M, Mpad);
      pack_feat_kernel<<<gr(64L * Mpad * 64), 256, 0, stream>>>(buf0, featp, M, Mpad);
      assign_kernel<<<wblk(64L * (Mpad / 16)), 256, 0, stream>>>(L1p, eclfp, L2p, assignT, M, Mpad);
      hf_kernel<<<wblk(64L * 8), 256, 0, stream>>>(assignT, featp, hfb, Mpad);
      edge_mix_kernel<<<64, 256, 0, stream>>>(hfb, emap, outp);
      scatter_kernel<<<wblk(64L * (Mpad / 16) * 4), 256, 0, stream>>>(L2p, outp, buf0, buf1, M, Mpad);
      ln_kernel<<<wblk(Rk), 256, 0, stream>>>(buf1, hln_s, hln_b, Rk);
      // combine
      avg_kernel<<<gr(EL), 256, 0, stream>>>(buf4, buf1, buf0, EL);
    }
    copy_lg_kernel<<<gr(64L * 325 * 64), 256, 0, stream>>>(buf0, lconc, gconc, Tk, k);
  }

  // ---- final projections + predictor ----
  lin_bf16_kernel<<<wblk(1300L * 4), 256, 0, stream>>>(lconc, localp, local_b, lfb, 1300, 384, 1);
  lin_bf16_kernel<<<wblk(1300L * 4), 256, 0, stream>>>(gconc, globalp, global_b, gfb, 1300, 384, 1);
  build_z_kernel<<<gr(64L * 325 * 160), 256, 0, stream>>>(lfb, gfb, target, zbf);
  lin_bf16_kernel<<<wblk(1300L * 4), 256, 0, stream>>>(zbf, p1p, p1_b, h1b, 1300, 160, 1);
  pred_kernel<<<gr(64L * 325 * 12), 256, 0, stream>>>(h1b, p2_w, p2_b, (float*)d_out);
}
